// SparseDiffusionAttention_62156766708189
// MI455X (gfx1250) — compile-verified, matrix operand order checked by reading, not run on hardware
//
#include <hip/hip_runtime.h>

typedef __attribute__((ext_vector_type(16))) _Float16 v16h;
typedef __attribute__((ext_vector_type(8)))  _Float16 v8h;
typedef __attribute__((ext_vector_type(4)))  _Float16 v4h;
typedef __attribute__((ext_vector_type(8)))  float    v8f;
typedef __attribute__((ext_vector_type(4)))  float    v4f;
typedef __attribute__((ext_vector_type(4)))  int      v4i;

#define DMODEL 1024
#define NHEADS 16
#define DK     64
#define RADIUS_ 8
#define WIN_    9
#define INV2T  (1.0f / 0.56f)   // 1/(2*T), T=0.28
#define KTILES (DMODEL / 32)    // 32
#define NTILES (DMODEL / 16)    // 64

#if defined(__AMDGCN__) && __has_builtin(__builtin_amdgcn_global_load_async_to_lds_b128) && __has_builtin(__builtin_amdgcn_s_wait_asynccnt)
#define SDA_ASYNC 1
#else
#define SDA_ASYNC 0
#endif

#if SDA_ASYNC
typedef __attribute__((address_space(1))) v4i gas_v4i;   // prints as "__device__"
typedef __attribute__((address_space(3))) v4i las_v4i;   // prints as "__shared__"
#endif

// ---------------------------------------------------------------------------
// Elementwise f32 -> f16 convert (vectorized 4-wide). Used once for x.
// ---------------------------------------------------------------------------
__global__ __launch_bounds__(256) void sda_cvt_f16(
    const float* __restrict__ in, _Float16* __restrict__ out, int n4)
{
  const int i = blockIdx.x * blockDim.x + threadIdx.x;
  if (i >= n4) return;
  const v4f v = ((const v4f*)in)[i];
  v4h h;
  #pragma unroll
  for (int j = 0; j < 4; ++j) h[j] = (_Float16)v[j];
  ((v4h*)out)[i] = h;
}

// ---------------------------------------------------------------------------
// Pack W [K=1024, N=1024] f32 row-major into fragment-major f16:
//   P[((kt*NTILES + nt)*32 + lane)*16 + e]  holds  W[k][n]  where
//   lane = (b<<4)|(n&15),  e = (a<<3)|c,  k_in_tile = 16a + 8b + c.
// ---------------------------------------------------------------------------
__global__ __launch_bounds__(256) void sda_pack_w(
    const float* __restrict__ W, _Float16* __restrict__ P)
{
  const int idx = blockIdx.x * blockDim.x + threadIdx.x; // 0 .. 1024*1024-1
  const int n = idx & (DMODEL - 1);
  const int k = idx >> 10;
  const int kt  = k >> 5;
  const int kin = k & 31;
  const int nt  = n >> 4;
  const int a = kin >> 4, b = (kin >> 3) & 1, c = kin & 7;
  const int lane = (b << 4) | (n & 15);
  const int e    = (a << 3) | c;
  P[(((size_t)kt * NTILES + nt) * 32 + lane) * 16 + e] = (_Float16)W[idx];
}

// ---------------------------------------------------------------------------
// GEMM: C[M,1024] = A[M,1024] * W. A f16 row-major, W pre-packed f16, C f32.
// Block = 8 waves stacked vertically: 256x64 output tile. The block's 4KB
// packed-B slab per K-step is staged into LDS (async copy, double-buffered)
// and shared by all 8 waves; each wave runs a 32x64 WMMA tile (8 WMMAs/step).
// ---------------------------------------------------------------------------
__global__ __launch_bounds__(256) void sda_gemm(
    const _Float16* __restrict__ A, const _Float16* __restrict__ P,
    float* __restrict__ C, int M)
{
  __shared__ _Float16 ldsB[2][4 * 512];      // 2 x 4KB double buffer

  const int tid  = threadIdx.x;
  const int lane = tid & 31;
  const int wib  = tid >> 5;                 // wave in block, 0..7
  const int bM   = blockIdx.x >> 4;          // row block (256 rows)
  const int bN   = blockIdx.x & 15;          // 64-col strip
  if (bM * 256 >= M) return;                 // block-uniform guard
  const int m0  = bM * 256 + wib * 32;
  const int nt0 = bN * 4;                    // first 16-col tile of strip
  const int r16 = lane & 15;
  const int kh  = (lane >> 4) << 3;          // lanes 16-31: K offset +8

  // Per-thread 16B slice of the 4KB B slab.
  const _Float16* gB0 = P + ((size_t)nt0 * 512) + (size_t)tid * 8;
  _Float16* lB[2] = { &ldsB[0][tid * 8], &ldsB[1][tid * 8] };

  v8f acc[2][4];
  #pragma unroll
  for (int hf = 0; hf < 2; ++hf)
    #pragma unroll
    for (int t = 0; t < 4; ++t)
      #pragma unroll
      for (int i = 0; i < 8; ++i) acc[hf][t][i] = 0.0f;

  const _Float16* arow0 = A + (size_t)(m0 + r16) * DMODEL + kh;
  const _Float16* arow1 = arow0 + (size_t)16 * DMODEL;

  // Stage K-step 0 into buffer 0.
#if SDA_ASYNC
  __builtin_amdgcn_global_load_async_to_lds_b128(
      (gas_v4i*)gB0, (las_v4i*)lB[0], 0, 0);
  __builtin_amdgcn_s_wait_asynccnt(0);
#else
  *(v16h*)lB[0] = *(const v16h*)gB0;
#endif
  __syncthreads();

  for (int kt = 0; kt < KTILES; ++kt) {
    const int buf = kt & 1;
    // Stage next K-step's B slab into the other buffer (overlaps with WMMAs).
    if (kt + 1 < KTILES) {
      const _Float16* gB = gB0 + (size_t)(kt + 1) * NTILES * 512;
#if SDA_ASYNC
      __builtin_amdgcn_global_load_async_to_lds_b128(
          (gas_v4i*)gB, (las_v4i*)lB[buf ^ 1], 0, 0);
#else
      *(v16h*)lB[buf ^ 1] = *(const v16h*)gB;
#endif
      __builtin_prefetch(arow0 + (kt + 1) * 32, 0, 1);   // A stream prefetch
    }

    // A fragments: two b128 loads each (K runs {kh..kh+7}, {kh+16..kh+23}).
    const v8h a0lo = *(const v8h*)(arow0 + kt * 32);
    const v8h a0hi = *(const v8h*)(arow0 + kt * 32 + 16);
    const v8h a1lo = *(const v8h*)(arow1 + kt * 32);
    const v8h a1hi = *(const v8h*)(arow1 + kt * 32 + 16);
    const v16h a0 = __builtin_shufflevector(a0lo, a0hi,
        0,1,2,3,4,5,6,7,8,9,10,11,12,13,14,15);
    const v16h a1 = __builtin_shufflevector(a1lo, a1hi,
        0,1,2,3,4,5,6,7,8,9,10,11,12,13,14,15);

    // B fragments from LDS: 32B contiguous per lane per tile.
    v16h bfr[4];
    #pragma unroll
    for (int t = 0; t < 4; ++t)
      bfr[t] = *(const v16h*)(&ldsB[buf][t * 512 + lane * 16]);

    #pragma unroll
    for (int t = 0; t < 4; ++t) {
      acc[0][t] = __builtin_amdgcn_wmma_f32_16x16x32_f16(
          false, a0, false, bfr[t], (short)0, acc[0][t], false, false);
      acc[1][t] = __builtin_amdgcn_wmma_f32_16x16x32_f16(
          false, a1, false, bfr[t], (short)0, acc[1][t], false, false);
    }

    // Next stage complete + all waves done reading before buffer reuse.
#if SDA_ASYNC
    __builtin_amdgcn_s_wait_asynccnt(0);
#endif
    __syncthreads();
  }

  // D store: VGPR r, lane l -> (M = (l>>4)*8 + r, N = l&15)
  const int mh = (lane >> 4) << 3;
  #pragma unroll
  for (int hf = 0; hf < 2; ++hf)
    #pragma unroll
    for (int t = 0; t < 4; ++t)
      #pragma unroll
      for (int r = 0; r < 8; ++r)
        C[(size_t)(m0 + hf * 16 + mh + r) * DMODEL + ((nt0 + t) * 16 + r16)]
            = acc[hf][t][r];
}

// ---------------------------------------------------------------------------
// Sliding-window attention: one thread per (b, h, s). Reads f32 Q/K/V,
// writes the attn matrix directly as f16 (feeds the final WMMA GEMM).
// ---------------------------------------------------------------------------
__global__ __launch_bounds__(256) void sda_attn(
    const float* __restrict__ Q, const float* __restrict__ K,
    const float* __restrict__ V, _Float16* __restrict__ O,
    int S, int total)
{
  const int idx = blockIdx.x * blockDim.x + threadIdx.x;
  if (idx >= total) return;
  const int s = idx % S;
  const int h = (idx / S) % NHEADS;
  const int b = idx / (S * NHEADS);
  const size_t base = ((size_t)b * S) * DMODEL + (size_t)h * DK;

  const float* q = Q + base + (size_t)s * DMODEL;
  float qr[DK];
  #pragma unroll
  for (int d = 0; d < DK; ++d) qr[d] = q[d];

  float wts[WIN_];
  float mx = -3.402823466e38f;
  #pragma unroll
  for (int w = 0; w < WIN_; ++w) {
    const int sp = s - RADIUS_ + w;
    if (sp < 0) { wts[w] = -3.402823466e38f; continue; }
    const float* kp = K + base + (size_t)sp * DMODEL;
    float dot = 0.0f;
    #pragma unroll
    for (int d = 0; d < DK; ++d) dot = fmaf(qr[d], kp[d], dot);
    wts[w] = dot * INV2T;
    mx = fmaxf(mx, wts[w]);
  }

  float denom = 0.0f;
  #pragma unroll
  for (int w = 0; w < WIN_; ++w) {
    const int sp = s - RADIUS_ + w;
    const float e = (sp < 0) ? 0.0f : __expf(wts[w] - mx);
    wts[w] = e;
    denom += e;
  }
  const float inv = 1.0f / denom;   // w = RADIUS (sp = s) always valid

  float out[DK];
  #pragma unroll
  for (int d = 0; d < DK; ++d) out[d] = 0.0f;
  #pragma unroll
  for (int w = 0; w < WIN_; ++w) {
    const int sp = s - RADIUS_ + w;
    if (sp < 0) continue;
    const float* vp = V + base + (size_t)sp * DMODEL;
    const float wt = wts[w];
    #pragma unroll
    for (int d = 0; d < DK; ++d) out[d] = fmaf(wt, vp[d], out[d]);
  }

  _Float16* op = O + base + (size_t)s * DMODEL;
  #pragma unroll
  for (int d = 0; d < DK; ++d) op[d] = (_Float16)(out[d] * inv);
}

// ---------------------------------------------------------------------------
extern "C" void kernel_launch(void* const* d_in, const int* in_sizes, int n_in,
                              void* d_out, int out_size, void* d_ws, size_t ws_size,
                              hipStream_t stream) {
  const float* x  = (const float*)d_in[0];
  const float* Wq = (const float*)d_in[1];
  const float* Wk = (const float*)d_in[2];
  const float* Wv = (const float*)d_in[3];
  const float* Wo = (const float*)d_in[4];

  const int M = in_sizes[0] / DMODEL;      // B*S = 8192
  const int S = 4096;                      // per reference
  const size_t MD = (size_t)M * DMODEL;
  const size_t WN = (size_t)DMODEL * DMODEL;

  // Workspace layout (all regions well-aligned by construction):
  float*    Qf = (float*)d_ws;
  float*    Kf = Qf + MD;
  float*    Vf = Kf + MD;
  _Float16* xh = (_Float16*)(Vf + MD);
  _Float16* At = xh + MD;
  _Float16* Pq = At + MD;
  _Float16* Pk = Pq + WN;
  _Float16* Pv = Pk + WN;
  _Float16* Po = Pv + WN;

  // 1) one-time conversions / packs
  const int n4 = (int)(MD / 4);
  sda_cvt_f16<<<(n4 + 255) / 256, 256, 0, stream>>>(x, xh, n4);
  const int packBlocks = (int)(WN / 256);
  sda_pack_w<<<packBlocks, 256, 0, stream>>>(Wq, Pq);
  sda_pack_w<<<packBlocks, 256, 0, stream>>>(Wk, Pk);
  sda_pack_w<<<packBlocks, 256, 0, stream>>>(Wv, Pv);
  sda_pack_w<<<packBlocks, 256, 0, stream>>>(Wo, Po);

  // 2) QKV projections (WMMA, LDS-staged B)
  const int blocks = (M / 256) * (DMODEL / 64);   // 256x64 tile per block
  sda_gemm<<<blocks, 256, 0, stream>>>(xh, Pq, Qf, M);
  sda_gemm<<<blocks, 256, 0, stream>>>(xh, Pk, Kf, M);
  sda_gemm<<<blocks, 256, 0, stream>>>(xh, Pv, Vf, M);

  // 3) sliding-window attention
  const int total = M * NHEADS;
  sda_attn<<<(total + 255) / 256, 256, 0, stream>>>(Qf, Kf, Vf, At, S, total);

  // 4) output projection (WMMA, LDS-staged B) -> d_out (f32)
  sda_gemm<<<blocks, 256, 0, stream>>>(At, Po, (float*)d_out, M);
}
